// CausalSelfAttention_82729660056378
// MI455X (gfx1250) — matrix operand order repaired and checked by reading, not compiled, not run on hardware
//
#include <hip/hip_runtime.h>

typedef _Float16 f16;
typedef __attribute__((ext_vector_type(16))) _Float16 v16h;
typedef __attribute__((ext_vector_type(8)))  _Float16 v8h;
typedef __attribute__((ext_vector_type(8)))  float    v8f;

#define B_SZ 4
#define T_SZ 2048
#define C_SZ 1024
#define H_SZ 16
#define D_SZ 64

// ---------------------------------------------------------------------------
// WMMA fragment helpers (CDNA5 wave32 layouts, CDNA5 ISA §7.12.2)
// ---------------------------------------------------------------------------
static __device__ __forceinline__ v16h make_v16(v8h lo, v8h hi) {
  v16h r;
#pragma unroll
  for (int i = 0; i < 8; ++i) { r[i] = lo[i]; r[i + 8] = hi[i]; }
  return r;
}

// A fragment, 16x32 f16 row-major tile: lane holds row m=l&15,
// halves 0..7 = K[koff..koff+7], halves 8..15 = K[16+koff..23+koff],
// koff = 8*(l>>4).  Two aligned 16B loads.
static __device__ __forceinline__ v16h load_a_frag(const f16* tile, int ld) {
  const int l = threadIdx.x & 31;
  const int m = l & 15;
  const int koff = (l >> 4) << 3;
  const f16* p = tile + m * ld + koff;
  return make_v16(*(const v8h*)p, *(const v8h*)(p + 16));
}

// B fragment, 32x16 f16 row-major tile: lane holds col n=l&15,
// halves 0..7 = K[koff..koff+7], halves 8..15 = K[16+koff..23+koff].
static __device__ __forceinline__ v16h load_b_frag(const f16* tile, int ld, int col0) {
  const int l = threadIdx.x & 31;
  const int koff = (l >> 4) << 3;
  const f16* p = tile + koff * ld + col0 + (l & 15);
  v16h r;
#pragma unroll
  for (int i = 0; i < 8; ++i) { r[i] = p[i * ld]; r[i + 8] = p[(16 + i) * ld]; }
  return r;
}

static __device__ __forceinline__ v8f wmma_f16(v16h a, v16h b, v8f c) {
  return __builtin_amdgcn_wmma_f32_16x16x32_f16(false, a, false, b, (short)0, c,
                                                false, false);
}

// ---------------------------------------------------------------------------
// fp32 -> f16 convert (x, w_attn, w_proj)
// ---------------------------------------------------------------------------
__global__ __launch_bounds__(256) void k_cvt(const float* __restrict__ in,
                                             f16* __restrict__ out, int n) {
  int i = blockIdx.x * blockDim.x + threadIdx.x;
  const int stride = gridDim.x * blockDim.x;
  for (; i < n; i += stride) out[i] = (f16)in[i];
}

// ---------------------------------------------------------------------------
// QKV GEMM: [8192,1024]f16 @ [1024,3072]f16 + bias -> Q/K/V [B,H,T,D] f16
// block = 256 thr = 8 waves (2x4); wave tile 32x64 (acc[2][4]); block tile
// 64x256; K-slab 32.  Each B fragment feeds 2 WMMAs (A-fragment reuse).
// ---------------------------------------------------------------------------
__global__ __launch_bounds__(256) void k_gemm_qkv(
    const f16* __restrict__ X, const f16* __restrict__ W,
    const float* __restrict__ bias, f16* __restrict__ Qo, f16* __restrict__ Ko,
    f16* __restrict__ Vo) {
  const int Kd = C_SZ;
  const int N = 3 * C_SZ;
  __shared__ __align__(16) f16 As[64 * 32];
  __shared__ __align__(16) f16 Bs[32 * 256];

  const int tid = threadIdx.x;
  const int lane = tid & 31;
  const int wave = tid >> 5;
  const int wm = wave >> 2;  // 0..1
  const int wn = wave & 3;   // 0..3
  const int m0 = blockIdx.y * 64;
  const int n0 = blockIdx.x * 256;

  v8f acc[2][4] = {};

  for (int k0 = 0; k0 < Kd; k0 += 32) {
    __syncthreads();
    {  // cooperative coalesced 128-bit staging into LDS
      const int r = tid >> 2, c = (tid & 3) << 3;
      *(v8h*)&As[r * 32 + c] = *(const v8h*)&X[(size_t)(m0 + r) * Kd + k0 + c];
#pragma unroll
      for (int rep = 0; rep < 4; ++rep) {
        const int rr = rep * 8 + (tid >> 5);
        const int cc = (tid & 31) << 3;
        *(v8h*)&Bs[rr * 256 + cc] =
            *(const v8h*)&W[(size_t)(k0 + rr) * N + n0 + cc];
      }
    }
    __syncthreads();
    if (k0 + 32 < Kd) {  // gfx1250 global_prefetch_b8 for next K-slab
      __builtin_prefetch(&X[(size_t)(m0 + (tid >> 2)) * Kd + k0 + 32], 0, 1);
      __builtin_prefetch(
          &W[(size_t)(k0 + 32 + (tid >> 5)) * N + n0 + ((tid & 31) << 3)], 0, 1);
    }
    v16h af[2];
    af[0] = load_a_frag(&As[(wm * 32 + 0) * 32], 32);
    af[1] = load_a_frag(&As[(wm * 32 + 16) * 32], 32);
#pragma unroll
    for (int nt = 0; nt < 4; ++nt) {
      const v16h bf = load_b_frag(Bs, 256, wn * 64 + nt * 16);
      acc[0][nt] = wmma_f16(af[0], bf, acc[0][nt]);
      acc[1][nt] = wmma_f16(af[1], bf, acc[1][nt]);
    }
  }

  // epilogue: +bias, scatter into [B,H,T,D] f16 per q/k/v section
  const int hi = lane >> 4, col = lane & 15;
#pragma unroll
  for (int nt = 0; nt < 4; ++nt) {
    const int n = n0 + wn * 64 + nt * 16 + col;
    const int sel = n >> 10;  // 0=q 1=k 2=v
    const int cch = n & (C_SZ - 1);
    const int h = cch >> 6, d = cch & 63;
    f16* dst = (sel == 0) ? Qo : (sel == 1) ? Ko : Vo;
    const float bv = bias[n];
#pragma unroll
    for (int mt = 0; mt < 2; ++mt) {
#pragma unroll
      for (int j = 0; j < 8; ++j) {
        const int row = m0 + wm * 32 + mt * 16 + j + 8 * hi;
        const int bb = row >> 11;  // / T
        const int t = row & (T_SZ - 1);
        const size_t idx = (((size_t)(bb * H_SZ + h) * T_SZ) + t) * D_SZ + d;
        dst[idx] = (f16)(acc[mt][nt][j] + bv);
      }
    }
  }
}

// ---------------------------------------------------------------------------
// Flash attention: block = 128 thr = 4 waves; wave owns 32 queries (two 16-row
// tiles), WG owns 128.  Per 32-key chunk: S=Q·K^T (8 WMMA, shared B frags),
// online softmax (f32 VALU + 16-lane shfl_xor row reductions), O+=P·V
// (8 WMMA, shared B frags).  K staged transposed in LDS; causal chunks a wave
// doesn't need are skipped (wave-uniform guard, barriers stay outside).
// ---------------------------------------------------------------------------
__global__ __launch_bounds__(128) void k_attn(const f16* __restrict__ Q,
                                              const f16* __restrict__ K,
                                              const f16* __restrict__ V,
                                              f16* __restrict__ Y) {
  __shared__ __align__(16) f16 Kt[64 * 32];        // [d][key] (transposed)
  __shared__ __align__(16) f16 Vs[32 * 64];        // [key][d]
  __shared__ __align__(16) f16 Ps[4][2][16 * 32];  // per-wave P tiles (rt0/rt1)

  const int tid = threadIdx.x;
  const int wave = tid >> 5, lane = tid & 31;
  const int hi = lane >> 4, col = lane & 15;
  const int bh = blockIdx.y;        // b*H + h
  const int q0 = blockIdx.x * 128;  // workgroup query base
  const int qw = q0 + wave * 32;    // wave query base (2 row tiles of 16)

  const size_t hoff = (size_t)bh * T_SZ * D_SZ;
  const f16* Qb = Q + hoff;
  const f16* Kb = K + hoff;
  const f16* Vb = V + hoff;

  // preload Q A-fragments: 2 row tiles x 2 d-chunks (constant over key loop)
  v16h qf[2][2];
  {
    const int m = lane & 15, koff = (lane >> 4) << 3;
#pragma unroll
    for (int rt = 0; rt < 2; ++rt) {
      const f16* qp = Qb + (size_t)(qw + rt * 16 + m) * D_SZ;
#pragma unroll
      for (int kk = 0; kk < 2; ++kk)
        qf[rt][kk] = make_v16(*(const v8h*)(qp + kk * 32 + koff),
                              *(const v8h*)(qp + kk * 32 + 16 + koff));
    }
  }

  v8f o[2][4] = {};
  float mrow[2][8], lrow[2][8];
#pragma unroll
  for (int rt = 0; rt < 2; ++rt)
#pragma unroll
    for (int j = 0; j < 8; ++j) { mrow[rt][j] = -1e30f; lrow[rt][j] = 0.0f; }

  const int kend = q0 + 128;  // causal: keys beyond WG's last query are dead
  for (int kc = 0; kc < kend; kc += 32) {
    __syncthreads();
    {  // stage K (transposed) and V for this 32-key chunk
      const int key = tid >> 2, dc = (tid & 3) << 4;
      const f16* kp = Kb + (size_t)(kc + key) * D_SZ + dc;
      const v8h ka = *(const v8h*)kp;
      const v8h kb2 = *(const v8h*)(kp + 8);
#pragma unroll
      for (int i = 0; i < 8; ++i) {
        Kt[(dc + i) * 32 + key] = ka[i];
        Kt[(dc + 8 + i) * 32 + key] = kb2[i];
      }
      const f16* vp = Vb + (size_t)(kc + key) * D_SZ + dc;
      *(v8h*)&Vs[key * 64 + dc] = *(const v8h*)vp;
      *(v8h*)&Vs[key * 64 + dc + 8] = *(const v8h*)(vp + 8);
    }
    __syncthreads();

    if (kc > qw + 31) continue;  // wave-uniform causal skip (no barriers below)

    // S = Q @ K^T : s[rt][nt], B fragments shared across the two row tiles
    v8f s[2][2] = {};
#pragma unroll
    for (int nt = 0; nt < 2; ++nt)
#pragma unroll
      for (int kk = 0; kk < 2; ++kk) {
        const v16h bf = load_b_frag(&Kt[kk * 32 * 32], 32, nt * 16);
        s[0][nt] = wmma_f16(qf[0][kk], bf, s[0][nt]);
        s[1][nt] = wmma_f16(qf[1][kk], bf, s[1][nt]);
      }

    // scale + causal mask + online softmax; row m=j+8*hi, col=lane&15
#pragma unroll
    for (int rt = 0; rt < 2; ++rt) {
#pragma unroll
      for (int j = 0; j < 8; ++j) {
        const int qg = qw + rt * 16 + j + 8 * hi;
        float s0 = s[rt][0][j] * 0.125f;  // 1/sqrt(64)
        float s1 = s[rt][1][j] * 0.125f;
        if (kc + col > qg) s0 = -1e30f;
        if (kc + 16 + col > qg) s1 = -1e30f;
        float mx = fmaxf(s0, s1);
#pragma unroll
        for (int off = 1; off < 16; off <<= 1)
          mx = fmaxf(mx, __shfl_xor(mx, off, 32));
        const float newm = fmaxf(mrow[rt][j], mx);
        const float alpha = __expf(mrow[rt][j] - newm);
        const float p0 = __expf(s0 - newm);
        const float p1 = __expf(s1 - newm);
        float ls = p0 + p1;
#pragma unroll
        for (int off = 1; off < 16; off <<= 1) ls += __shfl_xor(ls, off, 32);
        lrow[rt][j] = lrow[rt][j] * alpha + ls;
        mrow[rt][j] = newm;
#pragma unroll
        for (int dt = 0; dt < 4; ++dt) o[rt][dt][j] *= alpha;
        f16* pw = &Ps[wave][rt][(j + 8 * hi) * 32];  // C-layout -> A relayout
        pw[col] = (f16)p0;
        pw[16 + col] = (f16)p1;
      }
    }

    // O += P @ V : one k-step of 32 keys, 4 d-tiles, B fragments shared
    const v16h pf0 = load_a_frag(Ps[wave][0], 32);
    const v16h pf1 = load_a_frag(Ps[wave][1], 32);
#pragma unroll
    for (int dt = 0; dt < 4; ++dt) {
      const v16h bf = load_b_frag(Vs, 64, dt * 16);
      o[0][dt] = wmma_f16(pf0, bf, o[0][dt]);
      o[1][dt] = wmma_f16(pf1, bf, o[1][dt]);
    }
  }

  // normalize, write y as f16 in [B, T, C] (heads re-merged)
  const int b = bh >> 4, h = bh & 15;
#pragma unroll
  for (int rt = 0; rt < 2; ++rt) {
#pragma unroll
    for (int j = 0; j < 8; ++j) {
      const float inv = 1.0f / lrow[rt][j];
      const int qg = qw + rt * 16 + j + 8 * hi;
      const size_t base = ((size_t)(b * T_SZ + qg)) * C_SZ + h * D_SZ;
#pragma unroll
      for (int dt = 0; dt < 4; ++dt)
        Y[base + dt * 16 + col] = (f16)(o[rt][dt][j] * inv);
    }
  }
}

// ---------------------------------------------------------------------------
// Projection GEMM: y[8192,1024]f16 @ w_proj[1024,1024]f16 + bias -> fp32 out
// Same 64x256 block / 32x64 wave tiling as the QKV GEMM.
// ---------------------------------------------------------------------------
__global__ __launch_bounds__(256) void k_gemm_proj(
    const f16* __restrict__ Yh, const f16* __restrict__ W,
    const float* __restrict__ bias, float* __restrict__ Out) {
  const int Kd = C_SZ;
  const int N = C_SZ;
  __shared__ __align__(16) f16 As[64 * 32];
  __shared__ __align__(16) f16 Bs[32 * 256];

  const int tid = threadIdx.x;
  const int lane = tid & 31;
  const int wave = tid >> 5;
  const int wm = wave >> 2;
  const int wn = wave & 3;
  const int m0 = blockIdx.y * 64;
  const int n0 = blockIdx.x * 256;

  v8f acc[2][4] = {};

  for (int k0 = 0; k0 < Kd; k0 += 32) {
    __syncthreads();
    {
      const int r = tid >> 2, c = (tid & 3) << 3;
      *(v8h*)&As[r * 32 + c] = *(const v8h*)&Yh[(size_t)(m0 + r) * Kd + k0 + c];
#pragma unroll
      for (int rep = 0; rep < 4; ++rep) {
        const int rr = rep * 8 + (tid >> 5);
        const int cc = (tid & 31) << 3;
        *(v8h*)&Bs[rr * 256 + cc] =
            *(const v8h*)&W[(size_t)(k0 + rr) * N + n0 + cc];
      }
    }
    __syncthreads();
    if (k0 + 32 < Kd) {
      __builtin_prefetch(&Yh[(size_t)(m0 + (tid >> 2)) * Kd + k0 + 32], 0, 1);
      __builtin_prefetch(
          &W[(size_t)(k0 + 32 + (tid >> 5)) * N + n0 + ((tid & 31) << 3)], 0, 1);
    }
    v16h af[2];
    af[0] = load_a_frag(&As[(wm * 32 + 0) * 32], 32);
    af[1] = load_a_frag(&As[(wm * 32 + 16) * 32], 32);
#pragma unroll
    for (int nt = 0; nt < 4; ++nt) {
      const v16h bf = load_b_frag(Bs, 256, wn * 64 + nt * 16);
      acc[0][nt] = wmma_f16(af[0], bf, acc[0][nt]);
      acc[1][nt] = wmma_f16(af[1], bf, acc[1][nt]);
    }
  }

  const int hi = lane >> 4, col = lane & 15;
#pragma unroll
  for (int nt = 0; nt < 4; ++nt) {
    const int n = n0 + wn * 64 + nt * 16 + col;
    const float bv = bias[n];
#pragma unroll
    for (int mt = 0; mt < 2; ++mt) {
#pragma unroll
      for (int j = 0; j < 8; ++j) {
        const int row = m0 + wm * 32 + mt * 16 + j + 8 * hi;
        Out[(size_t)row * N + n] = acc[mt][nt][j] + bv;
      }
    }
  }
}

// ---------------------------------------------------------------------------
// Launch: convert -> QKV GEMM -> flash attention -> proj GEMM (one stream)
// ---------------------------------------------------------------------------
extern "C" void kernel_launch(void* const* d_in, const int* in_sizes, int n_in,
                              void* d_out, int out_size, void* d_ws,
                              size_t ws_size, hipStream_t stream) {
  const float* x = (const float*)d_in[0];
  const float* w_attn = (const float*)d_in[1];
  const float* b_attn = (const float*)d_in[2];
  const float* w_proj = (const float*)d_in[3];
  const float* b_proj = (const float*)d_in[4];
  float* out = (float*)d_out;

  const size_t nX = (size_t)B_SZ * T_SZ * C_SZ;  // 8388608
  const size_t nWa = (size_t)C_SZ * 3 * C_SZ;    // 3145728
  const size_t nWp = (size_t)C_SZ * C_SZ;        // 1048576

  f16* ws = (f16*)d_ws;
  f16* x16 = ws;   ws += nX;
  f16* wa16 = ws;  ws += nWa;
  f16* wp16 = ws;  ws += nWp;
  f16* q16 = ws;   ws += nX;
  f16* k16 = ws;   ws += nX;
  f16* v16 = ws;   ws += nX;
  f16* y16 = ws;   ws += nX;  // total ~92 MB of workspace

  k_cvt<<<1024, 256, 0, stream>>>(x, x16, (int)nX);
  k_cvt<<<512, 256, 0, stream>>>(w_attn, wa16, (int)nWa);
  k_cvt<<<256, 256, 0, stream>>>(w_proj, wp16, (int)nWp);

  // M=8192 -> 128 blocks of 64; N=3072 -> 12 blocks of 256
  k_gemm_qkv<<<dim3(12, 128), 256, 0, stream>>>(x16, wa16, b_attn, q16, k16,
                                                v16);
  // (T/128)=16 query-tiles x (B*H)=64 head-slices
  k_attn<<<dim3(16, 64), 128, 0, stream>>>(q16, k16, v16, y16);
  // N=1024 -> 4 blocks of 256
  k_gemm_proj<<<dim3(4, 128), 256, 0, stream>>>(y16, wp16, b_proj, out);

  (void)in_sizes; (void)n_in; (void)out_size; (void)ws_size;
}
